// AlphaStock_84112639525650
// MI455X (gfx1250) — matrix operand order repaired and checked by reading, not compiled, not run on hardware
//
#include <hip/hip_runtime.h>
#include <hip/hip_bf16.h>

typedef __attribute__((ext_vector_type(16))) _Float16 v16h;
typedef __attribute__((ext_vector_type(8)))  float    v8f;

#define DEV static __device__ __forceinline__

// problem dims
#define TT 96
#define DD 16
#define HH 128

// ---- workspace byte offsets (all 256-B aligned) ----
#define WS_WIH0   ((size_t)0)          // 512x32  f16 (K padded 16->32)
#define WS_WHH0   ((size_t)32768)      // 512x128 f16
#define WS_WIH1   ((size_t)163840)     // 512x128 f16
#define WS_WHH1   ((size_t)294912)     // 512x128 f16
#define WS_AW1    ((size_t)425984)     // 128x128 f16 (transposed: [n*128+k])
#define WS_AW2    ((size_t)458752)
#define WS_PROJ   ((size_t)491520)     // 64x128 f16
#define WS_WQ     ((size_t)507904)     // 64x64 f16
#define WS_WK     ((size_t)516096)
#define WS_WV     ((size_t)524288)
#define WS_FF1    ((size_t)532480)     // 128x64 f16
#define WS_FF2    ((size_t)548864)     // 64x128 f16
#define WS_SP1    ((size_t)565248)     // 32x64 f16
#define WS_B0     ((size_t)569344)     // 512 f32 (bih0+bhh0)
#define WS_B1     ((size_t)571392)
#define WS_GTAB   ((size_t)573440)     // 51 f32
#define WS_XP     ((size_t)573696)     // 4096x64 f16
#define WS_Q      ((size_t)1097984)    // 4096x64 f16
#define WS_K      ((size_t)1622272)    // 4096x64 f16
#define WS_VT     ((size_t)2146560)    // 32 x (64x128) f16, v transposed per batch
#define WS_H2     ((size_t)2670848)    // 4096x96x128 f16 history

DEV _Float16* wsh(void* ws, size_t off) { return (_Float16*)((char*)ws + off); }
DEV float*    wsf(void* ws, size_t off) { return (float*)((char*)ws + off); }

// ---- WMMA fragment helpers (CDNA5 16x16x32 f16 layouts) ----
// A (16x32, row-major src, row stride ld): lane m=l&15, hi=l>>4;
//   halves 0..15 -> k = hi*16 + i
DEV v16h loadA16(const _Float16* src, int ld) {
  int l = threadIdx.x & 31;
  int m = l & 15, hi = l >> 4;
  const _Float16* p = src + m * ld + hi * 16;
  v16h r;
#pragma unroll
  for (int i = 0; i < 8; ++i) r[i] = p[i];
#pragma unroll
  for (int i = 0; i < 8; ++i) r[8 + i] = p[8 + i];
  return r;
}
// B (32x16) stored K-fastest: arr[n*ld + k]; lane n=l&15, k = (l>>4)*16 + h
// -> 16 contiguous halves per lane (single 32B load)
DEV v16h loadB16(const _Float16* src, int ld) {
  int l = threadIdx.x & 31;
  int n = l & 15, hi = l >> 4;
  const _Float16* p = src + n * ld + hi * 16;
  v16h r;
#pragma unroll
  for (int i = 0; i < 16; ++i) r[i] = p[i];
  return r;
}
DEV v8f wmma32(v16h a, v16h b, v8f c) {
  return __builtin_amdgcn_wmma_f32_16x16x32_f16(false, a, false, b, (short)0, c, false, false);
}
// C/D (16x16 f32): lane n=l&15, row = r + 8*(l>>4)
DEV void storeC_f32(float* dst, int ld, v8f c) {
  int l = threadIdx.x & 31;
  int n = l & 15, hi = l >> 4;
#pragma unroll
  for (int r = 0; r < 8; ++r) dst[(r + hi * 8) * ld + n] = c[r];
}
DEV float sigf(float x) { return 1.f / (1.f + __expf(-x)); }

// ---- CDNA5 async global->LDS path (ASYNCcnt) ----
DEV void async_g2l_b128(void* lds_ptr, const void* gptr) {
  // the LDS address comes from the VDST operand; generic LDS pointer keeps wave-relative offset
  // in addr[31:0] (flat LDS aperture), so the low dword is the VDST payload.
  unsigned int lds_addr = (unsigned int)(unsigned long long)lds_ptr;
  asm volatile("global_load_async_to_lds_b128 %0, %1, off"
               :: "v"(lds_addr), "v"(gptr)
               : "memory");
}
DEV void wait_asynccnt_le2() {
#if __has_builtin(__builtin_amdgcn_s_wait_asynccnt)
  __builtin_amdgcn_s_wait_asynccnt(2);
#else
  asm volatile("s_wait_asynccnt 0x2" ::: "memory");
#endif
}
DEV void wait_asynccnt_0() {
#if __has_builtin(__builtin_amdgcn_s_wait_asynccnt)
  __builtin_amdgcn_s_wait_asynccnt(0);
#else
  asm volatile("s_wait_asynccnt 0x0" ::: "memory");
#endif
}

// ================= prep: f16 weight repack + rank-gate table =================
__global__ void prep_kernel(const float* Wih0, const float* Whh0, const float* Wih1,
                            const float* Whh1, const float* bih0, const float* bhh0,
                            const float* bih1, const float* bhh1, const float* aW1,
                            const float* aW2, const float* projW, const float* Wq,
                            const float* Wk, const float* Wv, const float* ff1W,
                            const float* ff2W, const float* sp1W, const float* remb,
                            const float* rw1W, const float* rw1b, const float* rw2W,
                            void* ws) {
  _Float16* wih0h = wsh(ws, WS_WIH0);
  _Float16* whh0h = wsh(ws, WS_WHH0);
  _Float16* wih1h = wsh(ws, WS_WIH1);
  _Float16* whh1h = wsh(ws, WS_WHH1);
  _Float16* aW1h  = wsh(ws, WS_AW1);
  _Float16* aW2h  = wsh(ws, WS_AW2);
  _Float16* projh = wsh(ws, WS_PROJ);
  _Float16* wqh   = wsh(ws, WS_WQ);
  _Float16* wkh   = wsh(ws, WS_WK);
  _Float16* wvh   = wsh(ws, WS_WV);
  _Float16* ff1h  = wsh(ws, WS_FF1);
  _Float16* ff2h  = wsh(ws, WS_FF2);
  _Float16* sp1h  = wsh(ws, WS_SP1);
  float* b0   = wsf(ws, WS_B0);
  float* b1   = wsf(ws, WS_B1);
  float* gtab = wsf(ws, WS_GTAB);

  int tid = blockIdx.x * blockDim.x + threadIdx.x;
  int nt  = gridDim.x * blockDim.x;
  for (int i = tid; i < 512 * 32; i += nt) {
    int n = i >> 5, k = i & 31;
    wih0h[i] = (k < 16) ? (_Float16)Wih0[n * 16 + k] : (_Float16)0.f;
  }
  for (int i = tid; i < 512 * 128; i += nt) {
    whh0h[i] = (_Float16)Whh0[i];
    wih1h[i] = (_Float16)Wih1[i];
    whh1h[i] = (_Float16)Whh1[i];
  }
  for (int i = tid; i < 512; i += nt) { b0[i] = bih0[i] + bhh0[i]; b1[i] = bih1[i] + bhh1[i]; }
  for (int i = tid; i < 128 * 128; i += nt) {
    int n = i >> 7, k = i & 127;
    aW1h[i] = (_Float16)aW1[k * 128 + n];
    aW2h[i] = (_Float16)aW2[k * 128 + n];
  }
  for (int i = tid; i < 64 * 128; i += nt) {        // (128,64) sources, transposed
    int n = i >> 7, k = i & 127;
    projh[i] = (_Float16)projW[k * 64 + n];
    ff2h[i]  = (_Float16)ff2W[k * 64 + n];
  }
  for (int i = tid; i < 64 * 64; i += nt) {
    int n = i >> 6, k = i & 63;
    wqh[i] = (_Float16)Wq[k * 64 + n];
    wkh[i] = (_Float16)Wk[k * 64 + n];
    wvh[i] = (_Float16)Wv[k * 64 + n];
  }
  for (int i = tid; i < 128 * 64; i += nt) {        // ff1 (64,128)
    int n = i >> 6, k = i & 63;
    ff1h[i] = (_Float16)ff1W[k * 128 + n];
  }
  for (int i = tid; i < 32 * 64; i += nt) {         // sp1 (64,32)
    int n = i >> 6, k = i & 63;
    sp1h[i] = (_Float16)sp1W[k * 32 + n];
  }
  for (int d = tid; d < 51; d += nt) {              // gate depends only on dist
    float s = 0.f;
    for (int j = 0; j < 16; ++j) {
      float a = rw1b[j];
      for (int e = 0; e < 32; ++e) a += remb[d * 32 + e] * rw1W[e * 16 + j];
      s += fmaxf(a, 0.f) * rw2W[j];
    }
    gtab[d] = 1.f / (1.f + __expf(-s));
  }
}

// ================= fused 2-layer LSTM, 16 sequences per WG =================
__global__ __launch_bounds__(128) void lstm_kernel(const float* __restrict__ x, void* ws) {
  const _Float16* wih0 = wsh(ws, WS_WIH0);
  const _Float16* whh0 = wsh(ws, WS_WHH0);
  const _Float16* wih1 = wsh(ws, WS_WIH1);
  const _Float16* whh1 = wsh(ws, WS_WHH1);
  const float* bias0 = wsf(ws, WS_B0);
  const float* bias1 = wsf(ws, WS_B1);
  _Float16* h2 = wsh(ws, WS_H2);

  __shared__ _Float16 sx[16 * 32];
  __shared__ _Float16 sh0[16 * 128];
  __shared__ _Float16 sh1[16 * 128];
  __shared__ float sc0[16 * 128];
  __shared__ float sc1[16 * 128];
  __shared__ float sg[16 * 512];

  const int tid = threadIdx.x, w = tid >> 5;
  const int seq0 = blockIdx.x * 16;

  for (int i = tid; i < 16 * 32; i += 128) sx[i] = (_Float16)0.f;
  for (int i = tid; i < 16 * 128; i += 128) {
    sh0[i] = (_Float16)0.f; sh1[i] = (_Float16)0.f;
    sc0[i] = 0.f; sc1[i] = 0.f;
  }
  __syncthreads();

  for (int t = 0; t < TT; ++t) {
    for (int i = tid; i < 16 * 16; i += 128) {
      int r = i >> 4, c = i & 15;
      sx[r * 32 + c] = (_Float16)x[((size_t)(seq0 + r) * TT + t) * DD + c];
    }
    __syncthreads();

    // layer0 gates: each wave owns one gate (128 cols = 8 N-tiles)
#pragma unroll
    for (int j = 0; j < 8; ++j) {
      int n0 = w * 128 + j * 16;
      v8f acc = {};
      acc = wmma32(loadA16(sx, 32), loadB16(wih0 + (size_t)n0 * 32, 32), acc);
#pragma unroll
      for (int kc = 0; kc < 4; ++kc)
        acc = wmma32(loadA16(sh0 + kc * 32, 128),
                     loadB16(whh0 + (size_t)n0 * 128 + kc * 32, 128), acc);
      storeC_f32(&sg[n0], 512, acc);
    }
    __syncthreads();

#pragma unroll 1
    for (int r = 0; r < 16; ++r) {
      int c = tid;
      float ig = sg[r * 512 + c] + bias0[c];
      float fg = sg[r * 512 + 128 + c] + bias0[128 + c];
      float gg = sg[r * 512 + 256 + c] + bias0[256 + c];
      float og = sg[r * 512 + 384 + c] + bias0[384 + c];
      float cc = sigf(fg) * sc0[r * 128 + c] + sigf(ig) * tanhf(gg);
      sc0[r * 128 + c] = cc;
      sh0[r * 128 + c] = (_Float16)(sigf(og) * tanhf(cc));
    }
    __syncthreads();

    // layer1 gates: K = h0(128) + h1_prev(128)
#pragma unroll
    for (int j = 0; j < 8; ++j) {
      int n0 = w * 128 + j * 16;
      v8f acc = {};
#pragma unroll
      for (int kc = 0; kc < 4; ++kc)
        acc = wmma32(loadA16(sh0 + kc * 32, 128),
                     loadB16(wih1 + (size_t)n0 * 128 + kc * 32, 128), acc);
#pragma unroll
      for (int kc = 0; kc < 4; ++kc)
        acc = wmma32(loadA16(sh1 + kc * 32, 128),
                     loadB16(whh1 + (size_t)n0 * 128 + kc * 32, 128), acc);
      storeC_f32(&sg[n0], 512, acc);
    }
    __syncthreads();

#pragma unroll 1
    for (int r = 0; r < 16; ++r) {
      int c = tid;
      float ig = sg[r * 512 + c] + bias1[c];
      float fg = sg[r * 512 + 128 + c] + bias1[128 + c];
      float gg = sg[r * 512 + 256 + c] + bias1[256 + c];
      float og = sg[r * 512 + 384 + c] + bias1[384 + c];
      float cc = sigf(fg) * sc1[r * 128 + c] + sigf(ig) * tanhf(gg);
      sc1[r * 128 + c] = cc;
      float hh = sigf(og) * tanhf(cc);
      sh1[r * 128 + c] = (_Float16)hh;
      h2[((size_t)(seq0 + r) * TT + t) * HH + c] = (_Float16)hh;
    }
    __syncthreads();
  }
}

// ========== temporal attention + LN1 + projection, 16 seqs per WG ==========
// h2_t tiles are double-buffered into LDS with GLOBAL_LOAD_ASYNC_TO_LDS_B128:
// issue tile t+1, s_wait_asynccnt<=2 (older tile resident, newest in flight),
// barrier, run the 8 WMMAs of tile t.
__global__ __launch_bounds__(128) void tattn_kernel(const float* __restrict__ attn_w,
                                                    const float* __restrict__ ln1_g,
                                                    const float* __restrict__ ln1_b,
                                                    const float* __restrict__ proj_b,
                                                    void* ws) {
  const _Float16* aW1 = wsh(ws, WS_AW1);
  const _Float16* aW2 = wsh(ws, WS_AW2);
  const _Float16* projW = wsh(ws, WS_PROJ);
  const _Float16* h2 = wsh(ws, WS_H2);
  _Float16* xp = wsh(ws, WS_XP);

  __shared__ _Float16 sHT[16 * 128];       // hT / final rep staging
  __shared__ _Float16 sDB[2][16 * 128];    // async double buffer for h2_t
  __shared__ float su[16 * 128];
  __shared__ float sz[16 * 128];
  __shared__ float salpha[16 * 96];
  __shared__ float sctx[16 * 128];
  __shared__ float smu[16], srv[16];

  const int tid = threadIdx.x, w = tid >> 5;
  const int seq0 = blockIdx.x * 16;

  // per-thread async issue of one 16x128 f16 tile (2 x B128 per thread)
  auto issue_tile = [&](int t, int buf) {
#pragma unroll
    for (int c = tid; c < 256; c += 128) {      // 256 chunks of 8 halves
      int r = c >> 4, col0 = (c & 15) * 8;
      const _Float16* g = h2 + ((size_t)(seq0 + r) * TT + t) * HH + col0;
      async_g2l_b128(&sDB[buf][c * 8], g);
    }
  };

  // stage hT synchronously
  for (int i = tid; i < 2048; i += 128) {
    int r = i >> 7, c = i & 127;
    sHT[i] = h2[((size_t)(seq0 + r) * TT + (TT - 1)) * HH + c];
  }
  __syncthreads();

  issue_tile(0, 0);  // prefetch t=0 while computing u

  // u = hT @ attn_W2
#pragma unroll
  for (int j = 0; j < 2; ++j) {
    int n0 = (w * 2 + j) * 16;
    v8f acc = {};
#pragma unroll
    for (int kc = 0; kc < 4; ++kc)
      acc = wmma32(loadA16(sHT + kc * 32, 128),
                   loadB16(aW2 + (size_t)n0 * 128 + kc * 32, 128), acc);
    storeC_f32(&su[n0], 128, acc);
  }
  __syncthreads();

  for (int t = 0; t < TT; ++t) {
    if (t + 1 < TT) {
      issue_tile(t + 1, (t + 1) & 1);
      wait_asynccnt_le2();    // tile t resident; tile t+1 may stay in flight
    } else {
      wait_asynccnt_0();
    }
    __syncthreads();

    const _Float16* sA = sDB[t & 1];
#pragma unroll
    for (int j = 0; j < 2; ++j) {
      int n0 = (w * 2 + j) * 16;
      v8f acc = {};
#pragma unroll
      for (int kc = 0; kc < 4; ++kc)
        acc = wmma32(loadA16(sA + kc * 32, 128),
                     loadB16(aW1 + (size_t)n0 * 128 + kc * 32, 128), acc);
      storeC_f32(&sz[n0], 128, acc);
    }
    __syncthreads();
    if (tid < 16) {
      float s = 0.f;
      for (int k = 0; k < 128; ++k)
        s += tanhf(sz[tid * 128 + k] + su[tid * 128 + k]) * attn_w[k];
      salpha[tid * 96 + t] = s;
    }
    __syncthreads();
  }

  if (tid < 16) {  // softmax over time
    float mx = -1e30f;
    for (int t = 0; t < TT; ++t) mx = fmaxf(mx, salpha[tid * 96 + t]);
    float sm = 0.f;
    for (int t = 0; t < TT; ++t) {
      float e = __expf(salpha[tid * 96 + t] - mx);
      salpha[tid * 96 + t] = e; sm += e;
    }
    float inv = 1.f / sm;
    for (int t = 0; t < TT; ++t) salpha[tid * 96 + t] *= inv;
  }
  __syncthreads();

  // ctx: thread owns one column across all 16 rows
  float acc[16];
#pragma unroll
  for (int r = 0; r < 16; ++r) acc[r] = 0.f;
  for (int t = 0; t < TT; ++t) {
#pragma unroll 1
    for (int r = 0; r < 16; ++r)
      acc[r] += salpha[r * 96 + t] * (float)h2[((size_t)(seq0 + r) * TT + t) * HH + tid];
  }
#pragma unroll
  for (int r = 0; r < 16; ++r) sctx[r * 128 + tid] = acc[r];
  __syncthreads();

  if (tid < 16) {  // LN stats
    float m = 0.f;
    for (int c = 0; c < 128; ++c) m += sctx[tid * 128 + c];
    m *= (1.f / 128.f);
    float v = 0.f;
    for (int c = 0; c < 128; ++c) { float d = sctx[tid * 128 + c] - m; v += d * d; }
    smu[tid] = m;
    srv[tid] = rsqrtf(v * (1.f / 128.f) + 1e-5f);
  }
  __syncthreads();
#pragma unroll 1
  for (int r = 0; r < 16; ++r)
    sHT[r * 128 + tid] =
        (_Float16)((sctx[r * 128 + tid] - smu[r]) * srv[r] * ln1_g[tid] + ln1_b[tid]);
  __syncthreads();

  {  // proj: one N-tile per wave
    int n0 = w * 16;
    v8f accp = {};
#pragma unroll
    for (int kc = 0; kc < 4; ++kc)
      accp = wmma32(loadA16(sHT + kc * 32, 128),
                    loadB16(projW + (size_t)n0 * 128 + kc * 32, 128), accp);
    int l = tid & 31, n = l & 15, hi = l >> 4;
#pragma unroll
    for (int r = 0; r < 8; ++r) {
      int m = r + hi * 8;
      xp[(size_t)(seq0 + m) * 64 + n0 + n] = (_Float16)(accp[r] + proj_b[n0 + n]);
    }
  }
}

// ========================= q/k/v projections =========================
__global__ __launch_bounds__(128) void qkv_kernel(const float* bq, const float* bk,
                                                  const float* bv, void* ws) {
  const _Float16* xp = wsh(ws, WS_XP);
  const _Float16* Wsel[3] = { wsh(ws, WS_WQ), wsh(ws, WS_WK), wsh(ws, WS_WV) };
  _Float16* q  = wsh(ws, WS_Q);
  _Float16* k  = wsh(ws, WS_K);
  _Float16* vT = wsh(ws, WS_VT);

  __shared__ _Float16 sX[16 * 64];
  const int tid = threadIdx.x, w = tid >> 5;
  const int row0 = blockIdx.x * 16;
  for (int i = tid; i < 1024; i += 128) sX[i] = xp[(size_t)row0 * 64 + i];
  __syncthreads();

  for (int job = w; job < 12; job += 4) {
    int o = job % 3, n0 = (job / 3) * 16;
    const _Float16* W = Wsel[o];
    const float* bias = (o == 0) ? bq : (o == 1) ? bk : bv;
    v8f acc = {};
#pragma unroll
    for (int kc = 0; kc < 2; ++kc)
      acc = wmma32(loadA16(sX + kc * 32, 64),
                   loadB16(W + (size_t)n0 * 64 + kc * 32, 64), acc);
    int l = tid & 31, n = l & 15, hi = l >> 4;
    int col = n0 + n;
#pragma unroll
    for (int r = 0; r < 8; ++r) {
      int row = row0 + r + hi * 8;
      float val = acc[r] + bias[col];
      if (o == 0) q[(size_t)row * 64 + col] = (_Float16)val;
      else if (o == 1) k[(size_t)row * 64 + col] = (_Float16)val;
      else vT[(size_t)(row >> 7) * 8192 + (size_t)col * 128 + (row & 127)] = (_Float16)val;
    }
  }
}

// ======== cross-asset attention + FF + LN2 + head, (batch, 16-row) per WG ========
__global__ __launch_bounds__(128) void attn2_kernel(const int* __restrict__ ranks,
                                                    const float* ff1_b, const float* ff2_b,
                                                    const float* ln2_g, const float* ln2_b,
                                                    const float* sp1_b, const float* sp2_W,
                                                    const float* sp2_b, float* out, void* ws) {
  const _Float16* q  = wsh(ws, WS_Q);
  const _Float16* kM = wsh(ws, WS_K);
  const _Float16* vT = wsh(ws, WS_VT);
  const _Float16* ff1h = wsh(ws, WS_FF1);
  const _Float16* ff2h = wsh(ws, WS_FF2);
  const _Float16* sp1h = wsh(ws, WS_SP1);
  const float* gtab = wsf(ws, WS_GTAB);

  const int b = blockIdx.x >> 3, m0 = (blockIdx.x & 7) * 16;
  const int tid = threadIdx.x, w = tid >> 5;

  __shared__ _Float16 sQ[16 * 64];
  __shared__ float ss[16 * 128];
  __shared__ _Float16 sP[16 * 128];
  __shared__ _Float16 sAo[16 * 64];
  __shared__ _Float16 sT1[16 * 128];
  __shared__ float sT2[16 * 64];
  __shared__ _Float16 sF[16 * 64];
  __shared__ _Float16 sS1[16 * 32];
  __shared__ int srk[128];
  __shared__ float sgt[51];

  for (int i = tid; i < 1024; i += 128) sQ[i] = q[((size_t)(b * 128 + m0)) * 64 + i];
  srk[tid] = ranks[b * 128 + tid];
  for (int i = tid; i < 51; i += 128) sgt[i] = gtab[i];
  __syncthreads();

  // scores = q k^T
#pragma unroll
  for (int j = 0; j < 2; ++j) {
    int n0 = (w * 2 + j) * 16;
    v8f acc = {};
#pragma unroll
    for (int kc = 0; kc < 2; ++kc)
      acc = wmma32(loadA16(sQ + kc * 32, 64),
                   loadB16(kM + ((size_t)(b * 128 + n0)) * 64 + kc * 32, 64), acc);
    storeC_f32(&ss[n0], 128, acc);
  }
  __syncthreads();

  if (tid < 16) {  // gated softmax -> probs f16
    int rn = srk[m0 + tid];
    float mx = -1e30f;
    for (int m = 0; m < 128; ++m) {
      int d = rn - srk[m]; d = d < 0 ? -d : d; if (d > 50) d = 50;
      float v = ss[tid * 128 + m] * 0.125f * sgt[d];
      ss[tid * 128 + m] = v;
      mx = fmaxf(mx, v);
    }
    float sm = 0.f;
    for (int m = 0; m < 128; ++m) {
      float e = __expf(ss[tid * 128 + m] - mx);
      ss[tid * 128 + m] = e; sm += e;
    }
    float inv = 1.f / sm;
    for (int m = 0; m < 128; ++m) sP[tid * 128 + m] = (_Float16)(ss[tid * 128 + m] * inv);
  }
  __syncthreads();

  {  // ao = P @ v (v pre-transposed)
    int n0 = w * 16;
    v8f acc = {};
#pragma unroll
    for (int kc = 0; kc < 4; ++kc)
      acc = wmma32(loadA16(sP + kc * 32, 128),
                   loadB16(vT + (size_t)b * 8192 + (size_t)n0 * 128 + kc * 32, 128), acc);
    int l = tid & 31, n = l & 15, hi = l >> 4;
#pragma unroll
    for (int r = 0; r < 8; ++r) sAo[(r + hi * 8) * 64 + n0 + n] = (_Float16)acc[r];
  }
  __syncthreads();

  // t1 = relu(ao @ ff1 + b1)
#pragma unroll
  for (int j = 0; j < 2; ++j) {
    int n0 = (w * 2 + j) * 16;
    v8f acc = {};
#pragma unroll
    for (int kc = 0; kc < 2; ++kc)
      acc = wmma32(loadA16(sAo + kc * 32, 64),
                   loadB16(ff1h + (size_t)n0 * 64 + kc * 32, 64), acc);
    int l = tid & 31, n = l & 15, hi = l >> 4;
#pragma unroll
    for (int r = 0; r < 8; ++r)
      sT1[(r + hi * 8) * 128 + n0 + n] = (_Float16)fmaxf(acc[r] + ff1_b[n0 + n], 0.f);
  }
  __syncthreads();

  {  // t2 = t1 @ ff2 + b2
    int n0 = w * 16;
    v8f acc = {};
#pragma unroll
    for (int kc = 0; kc < 4; ++kc)
      acc = wmma32(loadA16(sT1 + kc * 32, 128),
                   loadB16(ff2h + (size_t)n0 * 128 + kc * 32, 128), acc);
    int l = tid & 31, n = l & 15, hi = l >> 4;
#pragma unroll
    for (int r = 0; r < 8; ++r) sT2[(r + hi * 8) * 64 + n0 + n] = acc[r] + ff2_b[n0 + n];
  }
  __syncthreads();

  if (tid < 16) {  // LN2
    float m = 0.f;
    for (int c = 0; c < 64; ++c) m += sT2[tid * 64 + c];
    m *= (1.f / 64.f);
    float v = 0.f;
    for (int c = 0; c < 64; ++c) { float d = sT2[tid * 64 + c] - m; v += d * d; }
    float rinv = rsqrtf(v * (1.f / 64.f) + 1e-5f);
    for (int c = 0; c < 64; ++c)
      sF[tid * 64 + c] = (_Float16)((sT2[tid * 64 + c] - m) * rinv * ln2_g[c] + ln2_b[c]);
  }
  __syncthreads();

  if (w < 2) {  // s1 = relu(ff @ sp1 + b), wave-uniform branch
    int n0 = w * 16;
    v8f acc = {};
#pragma unroll
    for (int kc = 0; kc < 2; ++kc)
      acc = wmma32(loadA16(sF + kc * 32, 64),
                   loadB16(sp1h + (size_t)n0 * 64 + kc * 32, 64), acc);
    int l = tid & 31, n = l & 15, hi = l >> 4;
#pragma unroll
    for (int r = 0; r < 8; ++r)
      sS1[(r + hi * 8) * 32 + n0 + n] = (_Float16)fmaxf(acc[r] + sp1_b[n0 + n], 0.f);
  }
  __syncthreads();

  if (tid < 16) {  // sigmoid head
    float s = sp2_b[0];
    for (int j = 0; j < 32; ++j) s += (float)sS1[tid * 32 + j] * sp2_W[j];
    out[b * 128 + m0 + tid] = sigf(s);
  }
}

extern "C" void kernel_launch(void* const* d_in, const int* in_sizes, int n_in,
                              void* d_out, int out_size, void* d_ws, size_t ws_size,
                              hipStream_t stream) {
  (void)in_sizes; (void)n_in; (void)out_size; (void)ws_size;
  const float* x      = (const float*)d_in[0];
  const int*   ranks  = (const int*)d_in[1];
  const float* Wih0   = (const float*)d_in[2];
  const float* Whh0   = (const float*)d_in[3];
  const float* bih0   = (const float*)d_in[4];
  const float* bhh0   = (const float*)d_in[5];
  const float* Wih1   = (const float*)d_in[6];
  const float* Whh1   = (const float*)d_in[7];
  const float* bih1   = (const float*)d_in[8];
  const float* bhh1   = (const float*)d_in[9];
  const float* attn_w = (const float*)d_in[10];
  const float* aW1    = (const float*)d_in[11];
  const float* aW2    = (const float*)d_in[12];
  const float* ln1g   = (const float*)d_in[13];
  const float* ln1b   = (const float*)d_in[14];
  const float* projW  = (const float*)d_in[15];
  const float* projb  = (const float*)d_in[16];
  const float* Wq     = (const float*)d_in[17];
  const float* bq     = (const float*)d_in[18];
  const float* Wk     = (const float*)d_in[19];
  const float* bk     = (const float*)d_in[20];
  const float* Wv     = (const float*)d_in[21];
  const float* bv     = (const float*)d_in[22];
  const float* remb   = (const float*)d_in[23];
  const float* rw1W   = (const float*)d_in[24];
  const float* rw1b   = (const float*)d_in[25];
  const float* rw2W   = (const float*)d_in[26];
  const float* ff1W   = (const float*)d_in[27];
  const float* ff1b   = (const float*)d_in[28];
  const float* ff2W   = (const float*)d_in[29];
  const float* ff2b   = (const float*)d_in[30];
  const float* ln2g   = (const float*)d_in[31];
  const float* ln2b   = (const float*)d_in[32];
  const float* sp1W   = (const float*)d_in[33];
  const float* sp1b   = (const float*)d_in[34];
  const float* sp2W   = (const float*)d_in[35];
  const float* sp2b   = (const float*)d_in[36];

  prep_kernel<<<128, 256, 0, stream>>>(Wih0, Whh0, Wih1, Whh1, bih0, bhh0, bih1, bhh1,
                                       aW1, aW2, projW, Wq, Wk, Wv, ff1W, ff2W, sp1W,
                                       remb, rw1W, rw1b, rw2W, d_ws);
  lstm_kernel<<<256, 128, 0, stream>>>(x, d_ws);
  tattn_kernel<<<256, 128, 0, stream>>>(attn_w, ln1g, ln1b, projb, d_ws);
  qkv_kernel<<<256, 128, 0, stream>>>(bq, bk, bv, d_ws);
  attn2_kernel<<<256, 128, 0, stream>>>(ranks, ff1b, ff2b, ln2g, ln2b, sp1b,
                                        sp2W, sp2b, (float*)d_out, d_ws);
}
